// SwinBlock2D_81011673137636
// MI455X (gfx1250) — matrix operand order repaired and checked by reading, not compiled, not run on hardware
//
#include <hip/hip_runtime.h>
#include <hip/hip_bf16.h>
#include <math.h>

// ---------------- types ----------------
typedef __bf16 bf16_t;
typedef bf16_t bf16x16 __attribute__((ext_vector_type(16)));
typedef float  f32x8   __attribute__((ext_vector_type(8)));
typedef unsigned int u32x4 __attribute__((ext_vector_type(4)));
typedef unsigned int u32x8 __attribute__((ext_vector_type(8)));

union FragAB { uint4 u[2]; bf16x16 v; };
union Pack8  { unsigned short s[8]; uint4 v; };

__device__ __forceinline__ unsigned short f2bf(float f) {
    unsigned int u = __builtin_bit_cast(unsigned int, f);
    unsigned int r = u + 0x7FFFu + ((u >> 16) & 1u);
    return (unsigned short)(r >> 16);
}
__device__ __forceinline__ float bf2f(unsigned short h) {
    unsigned int u = ((unsigned int)h) << 16;
    return __builtin_bit_cast(float, u);
}
__device__ __forceinline__ f32x8 wmma_bf16(bf16x16 a, bf16x16 b, f32x8 c) {
    return __builtin_amdgcn_wmma_f32_16x16x32_bf16(false, a, false, b, (short)0, c, false, false);
}
__device__ __forceinline__ f32x8 zero8() {
    f32x8 z; for (int i = 0; i < 8; ++i) z[i] = 0.f; return z;
}

// problem constants
#define HH 64
#define WW_ 256
#define CC 256
#define NH 8
#define HD 32
#define NWIN 64          // tokens per window (8x8)
#define NWINDOWS 512     // B * (64/8) * (256/8)
#define MROWS 32768      // 512 * 64
#define LDA 40           // padded LDS row stride (halves): 32 data + 8 pad

// ---------------- TDM: DMA one 128x32 bf16 tile (row stride K halves) into LDS ----
// LDS layout: 128 rows of 32 halves (64B = 16 DW) + 16B pad (4 DW) => stride LDA=40.
// D# pad fields: pad_interval code 3 (=16 DW), pad_amount code 3 (=4 DW).
__device__ __forceinline__ void tdm_tile_load(const unsigned short* gsrc,
                                              unsigned int lds_off, int K) {
    unsigned long long ga = (unsigned long long)gsrc;
    u32x4 g0;
    g0[0] = 1u;                                            // count=1, user mode
    g0[1] = lds_off;                                       // lds_addr (bytes)
    g0[2] = (unsigned int)(ga & 0xFFFFFFFFu);              // global_addr[31:0]
    g0[3] = (unsigned int)((ga >> 32) & 0x01FFFFFFu)
          | (2u << 30);                                    // global_addr[56:32], type=2
    unsigned int Ku = (unsigned int)K;
    u32x8 g1;
    g1[0] = (1u << 16)                                     // data_size = 2 bytes
          | (1u << 20)                                     // pad_enable
          | (3u << 22)                                     // pad_interval: 16 DWORDs
          | (3u << 25);                                    // pad_amount: 4 DWORDs
    g1[1] = (Ku & 0xFFFFu) << 16;                          // tensor_dim0[15:0] @48
    g1[2] = ((Ku >> 16) & 0xFFFFu) | (128u << 16);         // tensor_dim0[31:16], tensor_dim1=128
    g1[3] = (32u << 16);                                   // tensor_dim1 hi=0, tile_dim0=32
    g1[4] = 128u;                                          // tile_dim1=128, tile_dim2=0
    g1[5] = Ku;                                            // tensor_dim0_stride[31:0]
    g1[6] = 0u;                                            // stride hi, dim1_stride lo
    g1[7] = 0u;
    asm volatile("tensor_load_to_lds %0, %1" :: "s"(g0), "s"(g1) : "memory");
}
__device__ __forceinline__ unsigned int lds_off_of(const void* p) {
    return (unsigned int)(unsigned long long)p;
}

// ---------------- weight transpose + bf16 convert: (K,N) f32 -> (N,K) bf16 --------
__global__ __launch_bounds__(256) void k_wtr(const float* __restrict__ w,
                                             unsigned short* __restrict__ wt,
                                             int K, int N) {
    int i = blockIdx.x * 256 + threadIdx.x;
    if (i >= K * N) return;
    int k = i / N, n = i % N;
    wt[(size_t)n * K + k] = f2bf(w[i]);
}

// ---------------- LayerNorm (one wave per 256-ch row), bf16 out -------------------
template <bool PERMUTE>
__global__ __launch_bounds__(256) void k_ln(const float* __restrict__ x,
                                            const float* __restrict__ g,
                                            const float* __restrict__ b,
                                            unsigned short* __restrict__ out) {
    int row  = blockIdx.x * 8 + (threadIdx.x >> 5);
    int lane = threadIdx.x & 31;
    size_t srow = (size_t)row;
    if (PERMUTE) {
        int win = row >> 6, n = row & 63;
        int bb = win >> 8, wq = win & 255;
        int wy = wq >> 5, wx = wq & 31;
        int hs = wy * 8 + (n >> 3), ws = wx * 8 + (n & 7);
        int ho = (hs + 4) & 63, wo = (ws + 4) & 255;
        srow = (size_t)bb * (HH * WW_) + ho * WW_ + wo;
    }
    const float* src = x + srow * CC + lane * 8;
    float4 p0 = *(const float4*)(src);
    float4 p1 = *(const float4*)(src + 4);
    float v[8] = {p0.x, p0.y, p0.z, p0.w, p1.x, p1.y, p1.z, p1.w};
    float s = 0.f, sq = 0.f;
    for (int i = 0; i < 8; ++i) { s += v[i]; sq += v[i] * v[i]; }
    for (int m = 16; m >= 1; m >>= 1) { s += __shfl_xor(s, m); sq += __shfl_xor(sq, m); }
    float mu = s * (1.f / CC);
    float var = sq * (1.f / CC) - mu * mu;
    float rs = rsqrtf(var + 1e-5f);
    Pack8 o;
    for (int i = 0; i < 8; ++i) {
        int c = lane * 8 + i;
        o.s[i] = f2bf((v[i] - mu) * rs * g[c] + b[c]);
    }
    *(uint4*)&out[(size_t)row * CC + lane * 8] = o.v;
}

// ---------------- tiled WMMA GEMM, TDM double-buffered, templated epilogue --------
// MODE 0: +bias, store bf16 row-major
// MODE 1: +bias, exact GELU, store bf16
// MODE 2: +bias, window_reverse + roll(+4,+4) permute, out[dst] = resid[dst] + v (f32)
// MODE 3: +bias, out[row*N+col] += v  (f32)
template <int MODE>
__global__ __launch_bounds__(256) void k_gemm(const unsigned short* __restrict__ A,
                                              const unsigned short* __restrict__ Bt,
                                              const float* __restrict__ bias,
                                              void* __restrict__ out,
                                              const float* __restrict__ resid,
                                              int M, int N, int K) {
    __shared__ __align__(16) unsigned short sA[2][128 * LDA];
    __shared__ __align__(16) unsigned short sB[2][128 * LDA];
    const int tid = threadIdx.x;
    const int lane = tid & 31, wid = tid >> 5;
    const int wm = wid >> 2, wn = wid & 3;        // wave tile: 64 rows x 32 cols
    const int lg = lane >> 4, lr = lane & 15;
    const int blockM = blockIdx.y * 128;
    const int blockN = blockIdx.x * 128;
    const unsigned short* Abase = A + (size_t)blockM * K;
    const unsigned short* Bbase = Bt + (size_t)blockN * K;

    f32x8 acc[4][2];
    for (int i = 0; i < 4; ++i) for (int j = 0; j < 2; ++j) acc[i][j] = zero8();

    if (wid == 0) {                               // prologue: DMA first tiles
        tdm_tile_load(Abase, lds_off_of(&sA[0][0]), K);
        tdm_tile_load(Bbase, lds_off_of(&sB[0][0]), K);
        __builtin_amdgcn_s_wait_tensorcnt(0);
    }
    __syncthreads();

    int buf = 0;
    for (int k0 = 0; k0 < K; k0 += 32) {
        if (wid == 0 && k0 + 32 < K) {            // prefetch next tiles via TDM
            tdm_tile_load(Abase + k0 + 32, lds_off_of(&sA[buf ^ 1][0]), K);
            tdm_tile_load(Bbase + k0 + 32, lds_off_of(&sB[buf ^ 1][0]), K);
        }
        FragAB afr[4], bfr[2];
#pragma unroll
        for (int mt = 0; mt < 4; ++mt) {
            int row = wm * 64 + mt * 16 + lr;
            afr[mt].u[0] = *(const uint4*)&sA[buf][row * LDA + lg * 8];
            afr[mt].u[1] = *(const uint4*)&sA[buf][row * LDA + 16 + lg * 8];
        }
#pragma unroll
        for (int nt = 0; nt < 2; ++nt) {
            int col = wn * 32 + nt * 16 + lr;
            bfr[nt].u[0] = *(const uint4*)&sB[buf][col * LDA + lg * 8];
            bfr[nt].u[1] = *(const uint4*)&sB[buf][col * LDA + 16 + lg * 8];
        }
#pragma unroll
        for (int mt = 0; mt < 4; ++mt)
#pragma unroll
            for (int nt = 0; nt < 2; ++nt)
                acc[mt][nt] = wmma_bf16(afr[mt].v, bfr[nt].v, acc[mt][nt]);
        if (wid == 0) __builtin_amdgcn_s_wait_tensorcnt(0);
        __syncthreads();
        buf ^= 1;
    }

#pragma unroll
    for (int mt = 0; mt < 4; ++mt)
#pragma unroll
        for (int nt = 0; nt < 2; ++nt) {
            int col = blockN + wn * 32 + nt * 16 + lr;
            float bv = bias[col];
#pragma unroll
            for (int r = 0; r < 8; ++r) {
                int row = blockM + wm * 64 + mt * 16 + r + lg * 8;
                float v = acc[mt][nt][r] + bv;
                if (MODE == 0) {
                    ((unsigned short*)out)[(size_t)row * N + col] = f2bf(v);
                } else if (MODE == 1) {
                    v = 0.5f * v * (1.f + erff(v * 0.70710678118654752f));
                    ((unsigned short*)out)[(size_t)row * N + col] = f2bf(v);
                } else if (MODE == 2) {
                    int win = row >> 6, n = row & 63;
                    int bb = win >> 8, wq = win & 255;
                    int wy = wq >> 5, wx = wq & 31;
                    int hs = wy * 8 + (n >> 3), ws = wx * 8 + (n & 7);
                    int ho = (hs + 4) & 63, wo = (ws + 4) & 255;
                    size_t dst = ((size_t)bb * (HH * WW_) + ho * WW_ + wo) * CC + col;
                    ((float*)out)[dst] = resid[dst] + v;
                } else {
                    ((float*)out)[(size_t)row * N + col] += v;
                }
            }
        }
}

// ---------------- windowed attention: 1 block per window, 1 wave per head --------
#define SLD 72
__global__ __launch_bounds__(256) void k_attn(const unsigned short* __restrict__ qkv, // (32768,768) bf16
                                              const float* __restrict__ rpb,          // (225,8)
                                              unsigned short* __restrict__ aout) {    // (32768,256) bf16
    __shared__ __align__(16) unsigned short sS[NH][NWIN * SLD]; // probs (wave-private per head)
    __shared__ __align__(16) unsigned short sV[NH][HD * SLD];   // V transposed (wave-private)
    __shared__ float sRpb[225 * NH];
    const int tid = threadIdx.x;
    for (int i = tid; i < 225 * NH; i += 256) sRpb[i] = rpb[i];
    const int win = blockIdx.x;
    const int h = tid >> 5, lane = tid & 31;
    const int lg = lane >> 4, lr = lane & 15;
    const unsigned short* base = qkv + (size_t)win * NWIN * (3 * CC) + h * HD;
    const unsigned short* Q  = base;
    const unsigned short* Kp = base + CC;
    const unsigned short* Vp = base + 2 * CC;

    // stage V^T into LDS (wave-private region)
    for (int i = lane; i < NWIN * HD; i += 32) {
        int kn = i >> 5, d = i & 31;
        sV[h][d * SLD + kn] = Vp[(size_t)kn * (3 * CC) + d];
    }
    __syncthreads();   // covers sRpb (the only cross-wave LDS data)

    // ---- S = Q K^T (K-dim = 32: single WMMA per 16x16 tile) ----
    FragAB qf[4];
#pragma unroll
    for (int mt = 0; mt < 4; ++mt) {
        int n = mt * 16 + lr;
        qf[mt].u[0] = *(const uint4*)&Q[(size_t)n * (3 * CC) + lg * 8];
        qf[mt].u[1] = *(const uint4*)&Q[(size_t)n * (3 * CC) + 16 + lg * 8];
    }
    f32x8 st[4][4];
#pragma unroll
    for (int nt = 0; nt < 4; ++nt) {
        FragAB kf;
        int kn = nt * 16 + lr;
        kf.u[0] = *(const uint4*)&Kp[(size_t)kn * (3 * CC) + lg * 8];
        kf.u[1] = *(const uint4*)&Kp[(size_t)kn * (3 * CC) + 16 + lg * 8];
#pragma unroll
        for (int mt = 0; mt < 4; ++mt)
            st[mt][nt] = wmma_bf16(qf[mt].v, kf.v, zero8());
    }

    // ---- scale + rel-pos bias + shift mask + in-register row softmax ----
    // Row identity = (mt, r, lg); its 64 cols live in nt=0..3 across lanes lr=0..15,
    // so the row reduction is 4 local values + shfl_xor over lane bits 0..3.
    const float scale = 0.17677669529663689f; // 32^-0.5
    const int wq = win & 255, wy = wq >> 5, wx = wq & 31;
#pragma unroll
    for (int mt = 0; mt < 4; ++mt) {
#pragma unroll
        for (int r = 0; r < 8; ++r) {
            int row = mt * 16 + r + lg * 8;
            int regr = ((wy == 7 && (row >> 3) >= 4) ? 2 : 0) + ((wx == 31 && (row & 7) >= 4) ? 1 : 0);
            float v[4];
            float mx = -1e30f;
#pragma unroll
            for (int nt = 0; nt < 4; ++nt) {
                int col = nt * 16 + lr;
                int ridx = ((row >> 3) - (col >> 3) + 7) * 15 + ((row & 7) - (col & 7) + 7);
                float bb = sRpb[ridx * NH + h];
                int regc = ((wy == 7 && (col >> 3) >= 4) ? 2 : 0) + ((wx == 31 && (col & 7) >= 4) ? 1 : 0);
                float mk = (regr != regc) ? -100.f : 0.f;
                float t = st[mt][nt][r] * scale + bb + mk;
                v[nt] = t;
                mx = fmaxf(mx, t);
            }
            for (int m = 8; m >= 1; m >>= 1) mx = fmaxf(mx, __shfl_xor(mx, m));
            float sum = 0.f;
#pragma unroll
            for (int nt = 0; nt < 4; ++nt) { v[nt] = __expf(v[nt] - mx); sum += v[nt]; }
            for (int m = 8; m >= 1; m >>= 1) sum += __shfl_xor(sum, m);
            float inv = 1.f / sum;
#pragma unroll
            for (int nt = 0; nt < 4; ++nt)
                sS[h][row * SLD + nt * 16 + lr] = f2bf(v[nt] * inv);
        }
    }
    // sS / sV are wave-private: same-wave DS ordering suffices, no barrier needed.

    // ---- O = P V (K-dim = 64: two WMMA steps) ----
    f32x8 oacc[4][2];
    for (int i = 0; i < 4; ++i) for (int j = 0; j < 2; ++j) oacc[i][j] = zero8();
#pragma unroll
    for (int kk = 0; kk < 2; ++kk) {
        FragAB pf[4], vf[2];
#pragma unroll
        for (int mt = 0; mt < 4; ++mt) {
            int row = mt * 16 + lr;
            pf[mt].u[0] = *(const uint4*)&sS[h][row * SLD + kk * 32 + lg * 8];
            pf[mt].u[1] = *(const uint4*)&sS[h][row * SLD + kk * 32 + 16 + lg * 8];
        }
#pragma unroll
        for (int nt = 0; nt < 2; ++nt) {
            int d = nt * 16 + lr;
            vf[nt].u[0] = *(const uint4*)&sV[h][d * SLD + kk * 32 + lg * 8];
            vf[nt].u[1] = *(const uint4*)&sV[h][d * SLD + kk * 32 + 16 + lg * 8];
        }
#pragma unroll
        for (int mt = 0; mt < 4; ++mt)
#pragma unroll
            for (int nt = 0; nt < 2; ++nt)
                oacc[mt][nt] = wmma_bf16(pf[mt].v, vf[nt].v, oacc[mt][nt]);
    }
#pragma unroll
    for (int mt = 0; mt < 4; ++mt)
#pragma unroll
        for (int nt = 0; nt < 2; ++nt)
#pragma unroll
            for (int r = 0; r < 8; ++r) {
                int n = mt * 16 + r + lg * 8;
                int c = h * HD + nt * 16 + lr;
                aout[((size_t)win * NWIN + n) * CC + c] = f2bf(oacc[mt][nt][r]);
            }
}

// ---------------- host launch ----------------
extern "C" void kernel_launch(void* const* d_in, const int* in_sizes, int n_in,
                              void* d_out, int out_size, void* d_ws, size_t ws_size,
                              hipStream_t stream) {
    (void)in_sizes; (void)n_in; (void)out_size; (void)ws_size;
    const float* x      = (const float*)d_in[0];
    const float* qkv_w  = (const float*)d_in[1];
    const float* qkv_b  = (const float*)d_in[2];
    const float* proj_w = (const float*)d_in[3];
    const float* proj_b = (const float*)d_in[4];
    const float* rpb    = (const float*)d_in[5];
    const float* n1w    = (const float*)d_in[6];
    const float* n1b    = (const float*)d_in[7];
    const float* n2w    = (const float*)d_in[8];
    const float* n2b    = (const float*)d_in[9];
    const float* w1     = (const float*)d_in[10];
    const float* b1     = (const float*)d_in[11];
    const float* w2     = (const float*)d_in[12];
    const float* b2     = (const float*)d_in[13];
    float* out = (float*)d_out;
    char* ws = (char*)d_ws;

    // workspace layout (bytes)
    unsigned short* WQT = (unsigned short*)(ws + 0);                 // 768x256 bf16
    unsigned short* WPT = (unsigned short*)(ws + 393216);            // 256x256
    unsigned short* W1T = (unsigned short*)(ws + 524288);            // 1024x256
    unsigned short* W2T = (unsigned short*)(ws + 1048576);           // 256x1024
    unsigned short* XN  = (unsigned short*)(ws + 2097152);           // 32768x256  (reused for YN)
    unsigned short* QKV = (unsigned short*)(ws + 18874368);          // 32768x768  (reused for HMID)
    unsigned short* AOUT= (unsigned short*)(ws + 69206016);          // 32768x256
    unsigned short* YN  = XN;
    unsigned short* HMID= QKV;

    k_wtr<<<(256 * 768 + 255) / 256, 256, 0, stream>>>(qkv_w, WQT, 256, 768);
    k_wtr<<<(256 * 256 + 255) / 256, 256, 0, stream>>>(proj_w, WPT, 256, 256);
    k_wtr<<<(256 * 1024 + 255) / 256, 256, 0, stream>>>(w1, W1T, 256, 1024);
    k_wtr<<<(1024 * 256 + 255) / 256, 256, 0, stream>>>(w2, W2T, 1024, 256);

    k_ln<true><<<MROWS / 8, 256, 0, stream>>>(x, n1w, n1b, XN);
    k_gemm<0><<<dim3(6, 256), 256, 0, stream>>>(XN, WQT, qkv_b, QKV, nullptr, MROWS, 768, 256);
    k_attn<<<NWINDOWS, 256, 0, stream>>>(QKV, rpb, AOUT);
    k_gemm<2><<<dim3(2, 256), 256, 0, stream>>>(AOUT, WPT, proj_b, out, x, MROWS, 256, 256);
    k_ln<false><<<MROWS / 8, 256, 0, stream>>>(out, n2w, n2b, YN);
    k_gemm<1><<<dim3(8, 256), 256, 0, stream>>>(YN, W1T, b1, HMID, nullptr, MROWS, 1024, 256);
    k_gemm<3><<<dim3(2, 256), 256, 0, stream>>>(HMID, W2T, b2, out, nullptr, MROWS, 256, 1024);
}
